// ForgetAttnLSTMArchived_88433376625190
// MI455X (gfx1250) — matrix operand order repaired and checked by reading, not compiled, not run on hardware
//
#include <hip/hip_runtime.h>
#include <math.h>

// ---------------------------------------------------------------------------
// Problem constants (from reference)
// ---------------------------------------------------------------------------
constexpr int kHS = 512;
constexpr int kL  = 4;
constexpr int kT  = 128;
constexpr int kM  = 2048;
constexpr int kFF = 2048;
constexpr int kNH = 4;
constexpr int kDH = kHS / kNH;   // 128

// ---------------------------------------------------------------------------
// CDNA5 WMMA types
// ---------------------------------------------------------------------------
typedef __attribute__((ext_vector_type(16))) __bf16 v16bf;
typedef __attribute__((ext_vector_type(8)))  float  v8f;

struct U4 { unsigned int a, b, c, d; };   // POD 16-byte chunk (b128 loads)

union BF16Frag {
    v16bf          v;
    unsigned short u[16];
    U4             q[2];
};

__device__ __forceinline__ unsigned short f2bf(float f) {
    unsigned int u = __float_as_uint(f);
    unsigned int r = (u + 0x7FFFu + ((u >> 16) & 1u)) >> 16;   // RNE
    return (unsigned short)r;
}

__device__ __forceinline__ float sigm(float x) { return 1.0f / (1.0f + __expf(-x)); }

// ---------------------------------------------------------------------------
// Kernel 1: pack historical_states / Wk / Wv to bf16, zero h/c, init barrier
// ---------------------------------------------------------------------------
__global__ void prep_kernel(const float* __restrict__ hist,
                            const float* __restrict__ ca_Win,
                            unsigned short* __restrict__ histb,
                            unsigned short* __restrict__ wkb,
                            unsigned short* __restrict__ wvb,
                            float* __restrict__ hbuf,
                            float* __restrict__ cbuf,
                            int* __restrict__ bar) {
    const size_t stride = (size_t)gridDim.x * blockDim.x;
    const size_t NH_TOT = (size_t)kL * kM * kHS;     // 4,194,304
    const size_t NW_TOT = (size_t)kL * kHS * kHS;    // 1,048,576
    for (size_t n = (size_t)blockIdx.x * blockDim.x + threadIdx.x; n < NH_TOT; n += stride) {
        size_t l = n / ((size_t)kM * kHS);
        size_t r = n - l * (size_t)kM * kHS;
        size_t m = r / kHS, e = r - m * kHS;
        histb[n] = f2bf(hist[(m * kL + l) * kHS + e]);
        if (n < NW_TOT) {
            size_t l2 = n / ((size_t)kHS * kHS);
            size_t rr = n - l2 * (size_t)kHS * kHS;
            size_t f = rr / kHS, e2 = rr - f * kHS;
            const float* base = ca_Win + l2 * (size_t)3 * kHS * kHS;
            wkb[n] = f2bf(base[(kHS     + f) * kHS + e2]);   // ca_Win[l, hs+f,  e]
            wvb[n] = f2bf(base[(2 * kHS + f) * kHS + e2]);   // ca_Win[l, 2hs+f, e]
        }
        if (n < (size_t)kL * kHS) { hbuf[n] = 0.0f; cbuf[n] = 0.0f; }
        if (n == 0) { bar[0] = 0; bar[1] = 0; }
    }
}

// ---------------------------------------------------------------------------
// Kernel 2: K/V pre-projection GEMM with v_wmma_f32_16x16x32_bf16
//   K[l,m,f]  = sum_e hist[m,l,e] * Wk[l,f,e] + bk[l,f]
//   Vt[l,f,m] = sum_e hist[m,l,e] * Wv[l,f,e] + bv[l,f]   (transposed)
// ---------------------------------------------------------------------------
__global__ __launch_bounds__(256) void kv_gemm_kernel(
        const unsigned short* __restrict__ histb,
        const unsigned short* __restrict__ wkb,
        const unsigned short* __restrict__ wvb,
        const float* __restrict__ ca_bin,
        float* __restrict__ Kout,
        float* __restrict__ Vt) {
    const int lane = threadIdx.x & 31;
    const int wave = (int)((blockIdx.x * blockDim.x + threadIdx.x) >> 5);
    const int ntm = kM / 16, ntn = kHS / 16;          // 128 x 32 tiles per layer
    if (wave >= kL * ntm * ntn) return;               // wave-uniform

    const int l   = wave / (ntm * ntn);
    const int rem = wave - l * (ntm * ntn);
    const int m0  = (rem / ntn) * 16;
    const int f0  = (rem % ntn) * 16;

    const unsigned short* A  = histb + (size_t)l * kM * kHS;
    const unsigned short* WK = wkb   + (size_t)l * kHS * kHS;
    const unsigned short* WV = wvb   + (size_t)l * kHS * kHS;

    const int mrow  = m0 + (lane & 15);
    const int khalf = lane >> 4;
    const int fcol  = f0 + (lane & 15);

    v8f ck, cv;
    #pragma unroll
    for (int i = 0; i < 8; ++i) { ck[i] = 0.0f; cv[i] = 0.0f; }

    const unsigned short* Arow = A  + (size_t)mrow * kHS;
    const unsigned short* Krow = WK + (size_t)fcol * kHS;
    const unsigned short* Vrow = WV + (size_t)fcol * kHS;

    for (int e0 = 0; e0 < kHS; e0 += 32) {
        BF16Frag a, bk, bv;
        // A (16x32 bf16): lanes 0-15 rows M=0..15, K=0..15; lanes 16-31 K=16..31
        a.q[0] = *(const U4*)(Arow + e0 + 16 * khalf);
        a.q[1] = *(const U4*)(Arow + e0 + 16 * khalf + 8);
        // B (32x16 bf16): lane = col N, lanes 0-15 hold K=0..15, lanes 16-31 K=16..31
        bk.q[0] = *(const U4*)(Krow + e0 + 16 * khalf);
        bk.q[1] = *(const U4*)(Krow + e0 + 16 * khalf + 8);
        bv.q[0] = *(const U4*)(Vrow + e0 + 16 * khalf);
        bv.q[1] = *(const U4*)(Vrow + e0 + 16 * khalf + 8);
        if (e0 + 32 < kHS) {
            __builtin_prefetch(Arow + e0 + 32, 0, 3);   // global_prefetch_b8, near-scope
        }
        ck = __builtin_amdgcn_wmma_f32_16x16x32_bf16(false, a.v, false, bk.v,
                                                     (short)0, ck, false, false);
        cv = __builtin_amdgcn_wmma_f32_16x16x32_bf16(false, a.v, false, bv.v,
                                                     (short)0, cv, false, false);
    }

    const float bkb = ca_bin[(size_t)l * 3 * kHS + kHS     + fcol];
    const float bvb = ca_bin[(size_t)l * 3 * kHS + 2 * kHS + fcol];
    // C/D layout: VGPR v -> row v + 8*khalf, col lane&15
    #pragma unroll
    for (int v = 0; v < 8; ++v) {
        const int row = m0 + v + 8 * khalf;
        Kout[((size_t)l * kM + row) * kHS + fcol] = ck[v] + bkb;
        Vt[((size_t)l * kHS + fcol) * kM + row]   = cv[v] + bvb;
    }
}

// ---------------------------------------------------------------------------
// Kernel 3: persistent recurrence over (t, layer) with device-wide barriers.
// 64 blocks x 256 threads = 512 waves; wave-per-row dot products. The shared
// (broadcast) vector operand of each matvec phase except P7 is staged into LDS with
// GLOBAL_LOAD_ASYNC_TO_LDS_B32 (ASYNCcnt) so weight rows are the only global
// traffic in the inner loops.
// ---------------------------------------------------------------------------
struct RecArgs {
    const float *x, *Wx, *bx, *Wh, *bh;
    const float *sa_Win, *sa_bin, *sa_Wo, *sa_bo;
    const float *ca_Win, *ca_bin, *ca_Wo, *ca_bo;
    const float *ff_W1, *ff_b1, *ff_W2, *ff_b2;
    const float *ln_g, *ln_b;
    const float *Kf, *Vt;
    float *z, *f0, *vsa, *r1, *x1, *q, *sc, *ctx, *r2, *x2, *ffa, *r3, *fdec, *xinb;
    float *h, *c;
    float *out;
    int   *bar;
};

__device__ __forceinline__ float wdot(const float* __restrict__ w,
                                      const float* __restrict__ v, int n) {
    const int lane = threadIdx.x & 31;
    float acc = 0.0f;
    for (int j = lane; j < n; j += 32) acc = fmaf(w[j], v[j], acc);
    #pragma unroll
    for (int o = 16; o > 0; o >>= 1) acc += __shfl_down(acc, o, 32);
    return acc;   // valid on lane 0
}

// Stage n floats from global memory into LDS using the async copy engine.
// Each enabled lane copies one dword per issued instruction:
//   global dword -> LDS offset        (tracked by ASYNCcnt)
__device__ __forceinline__ void stage_async(const float* __restrict__ g,
                                            float* sh, int n) {
    for (int j = threadIdx.x; j < n; j += 256) {
        unsigned lds_off = (unsigned)(size_t)(sh + j);          // flat->LDS: low 32 bits
        unsigned long long ga = (unsigned long long)(size_t)(g + j);
        asm volatile("global_load_async_to_lds_b32 %0, %1, off"
                     :: "v"(lds_off), "v"(ga) : "memory");
    }
    asm volatile("s_wait_asynccnt 0" ::: "memory");
    __syncthreads();
}

__device__ __forceinline__ void gbar(int* bar) {
    __syncthreads();
    if (threadIdx.x == 0) {
        volatile int* vcnt   = bar;
        volatile int* vsense = bar + 1;
        const int s = *vsense;
        __threadfence();
        if (atomicAdd(bar, 1) == (int)gridDim.x - 1) {
            *vcnt = 0;
            __threadfence();
            atomicAdd(bar + 1, 1);
        } else {
            while (*vsense == s) { __builtin_amdgcn_s_sleep(2); }
        }
    }
    __syncthreads();
    __threadfence();
}

// LayerNorm over kHS values; call from block 0 only (256 threads).
__device__ __forceinline__ void block_ln(const float* __restrict__ in,
                                         float* __restrict__ dst,
                                         const float* __restrict__ g,
                                         const float* __restrict__ b,
                                         float* red) {
    const int tid = threadIdx.x;
    float s = 0.0f;
    for (int j = tid; j < kHS; j += 256) s += in[j];
    red[tid] = s; __syncthreads();
    for (int o = 128; o > 0; o >>= 1) { if (tid < o) red[tid] += red[tid + o]; __syncthreads(); }
    const float mean = red[0] * (1.0f / kHS); __syncthreads();
    float vs = 0.0f;
    for (int j = tid; j < kHS; j += 256) { float d = in[j] - mean; vs = fmaf(d, d, vs); }
    red[tid] = vs; __syncthreads();
    for (int o = 128; o > 0; o >>= 1) { if (tid < o) red[tid] += red[tid + o]; __syncthreads(); }
    const float rstd = rsqrtf(red[0] * (1.0f / kHS) + 1e-5f); __syncthreads();
    for (int j = tid; j < kHS; j += 256) dst[j] = (in[j] - mean) * rstd * g[j] + b[j];
    __syncthreads();
}

__device__ __forceinline__ void softmax_row(float* __restrict__ s, int n, float* red) {
    const int tid = threadIdx.x;
    float mx = -1e30f;
    for (int j = tid; j < n; j += 256) mx = fmaxf(mx, s[j]);
    red[tid] = mx; __syncthreads();
    for (int o = 128; o > 0; o >>= 1) { if (tid < o) red[tid] = fmaxf(red[tid], red[tid + o]); __syncthreads(); }
    mx = red[0]; __syncthreads();
    float sum = 0.0f;
    for (int j = tid; j < n; j += 256) { float e = __expf(s[j] - mx); s[j] = e; sum += e; }
    red[tid] = sum; __syncthreads();
    for (int o = 128; o > 0; o >>= 1) { if (tid < o) red[tid] += red[tid + o]; __syncthreads(); }
    const float inv = 1.0f / red[0]; __syncthreads();
    for (int j = tid; j < n; j += 256) s[j] *= inv;
}

__global__ __launch_bounds__(256) void recurrence_kernel(RecArgs A) {
    __shared__ float red[256];
    __shared__ float sh[kFF];            // 8 KB staging buffer (max operand = ffa)
    const int tid  = threadIdx.x;
    const int lane = tid & 31;
    const int wid  = (int)((blockIdx.x * blockDim.x + tid) >> 5);
    const int nw   = (int)((gridDim.x * blockDim.x) >> 5);
    const float qscale = 0.08838834764831845f;   // 1/sqrt(128)

    for (int t = 0; t < kT; ++t) {
        for (int l = 0; l < kL; ++l) {
            const float* xin = (l == 0) ? (A.x + (size_t)t * kHS) : A.xinb;
            const float* hv  = A.h + (size_t)l * kHS;

            // P0: z = Wx@xin + bx + Wh@h + bh ; f0 = sigmoid(z1)
            stage_async(xin, sh, kHS);
            stage_async(hv, sh + kHS, kHS);
            for (int r = wid; r < 3 * kHS; r += nw) {
                const int g = r / kHS, i = r - g * kHS;
                const size_t ro = (((size_t)l * 3 + g) * kHS + i) * kHS;
                float acc = wdot(A.Wx + ro, sh, kHS) + wdot(A.Wh + ro, sh + kHS, kHS);
                if (lane == 0) {
                    const size_t bo = ((size_t)l * 3 + g) * kHS + i;
                    acc += A.bx[bo] + A.bh[bo];
                    A.z[r] = acc;
                    if (g == 1) A.f0[i] = sigm(acc);
                }
            }
            gbar(A.bar);

            // P1: v_sa = sa_Win[l, 2hs:] @ f0 + sa_bin[l, 2hs:]
            stage_async(A.f0, sh, kHS);
            for (int r = wid; r < kHS; r += nw) {
                const size_t ro = ((size_t)l * 3 * kHS + 2 * kHS + r) * kHS;
                float acc = wdot(A.sa_Win + ro, sh, kHS);
                if (lane == 0) A.vsa[r] = acc + A.sa_bin[(size_t)l * 3 * kHS + 2 * kHS + r];
            }
            gbar(A.bar);

            // P2: r1 = f0 + sa_Wo @ v_sa + sa_bo
            stage_async(A.vsa, sh, kHS);
            for (int r = wid; r < kHS; r += nw) {
                float acc = wdot(A.sa_Wo + ((size_t)l * kHS + r) * kHS, sh, kHS);
                if (lane == 0) A.r1[r] = A.f0[r] + acc + A.sa_bo[(size_t)l * kHS + r];
            }
            gbar(A.bar);

            // P3: x1 = LN(r1; ln[l,0])
            if (blockIdx.x == 0)
                block_ln(A.r1, A.x1, A.ln_g + ((size_t)l * 3 + 0) * kHS,
                                     A.ln_b + ((size_t)l * 3 + 0) * kHS, red);
            gbar(A.bar);

            // P4: q = ca_Win[l, :hs] @ x1 + ca_bin[l, :hs]
            stage_async(A.x1, sh, kHS);
            for (int r = wid; r < kHS; r += nw) {
                float acc = wdot(A.ca_Win + ((size_t)l * 3 * kHS + r) * kHS, sh, kHS);
                if (lane == 0) A.q[r] = acc + A.ca_bin[(size_t)l * 3 * kHS + r];
            }
            gbar(A.bar);

            // P5: scores[h,m] = scale * q[h,:] . K[l,m,h,:]
            stage_async(A.q, sh, kHS);
            for (int r = wid; r < kNH * kM; r += nw) {
                const int hh = r / kM, m = r - hh * kM;
                const float* krow = A.Kf + ((size_t)l * kM + m) * kHS + hh * kDH;
                float acc = wdot(krow, sh + hh * kDH, kDH);
                if (lane == 0) A.sc[r] = acc * qscale;
            }
            gbar(A.bar);

            // P6: softmax over m for each head (blocks 0..NH-1)
            if (blockIdx.x < kNH) softmax_row(A.sc + (size_t)blockIdx.x * kM, kM, red);
            gbar(A.bar);

            // P7: ctx[f] = att[h,:] . Vt[l,f,:]  (att = 32KB, streamed from L2)
            for (int r = wid; r < kHS; r += nw) {
                const int hh = r / kDH;
                float acc = wdot(A.Vt + ((size_t)l * kHS + r) * kM, A.sc + (size_t)hh * kM, kM);
                if (lane == 0) A.ctx[r] = acc;
            }
            gbar(A.bar);

            // P8: r2 = x1 + ca_Wo @ ctx + ca_bo
            stage_async(A.ctx, sh, kHS);
            for (int r = wid; r < kHS; r += nw) {
                float acc = wdot(A.ca_Wo + ((size_t)l * kHS + r) * kHS, sh, kHS);
                if (lane == 0) A.r2[r] = A.x1[r] + acc + A.ca_bo[(size_t)l * kHS + r];
            }
            gbar(A.bar);

            // P9: x2 = LN(r2; ln[l,1])
            if (blockIdx.x == 0)
                block_ln(A.r2, A.x2, A.ln_g + ((size_t)l * 3 + 1) * kHS,
                                     A.ln_b + ((size_t)l * 3 + 1) * kHS, red);
            gbar(A.bar);

            // P10: ffa = relu(ff_W1 @ x2 + ff_b1)
            stage_async(A.x2, sh, kHS);
            for (int r = wid; r < kFF; r += nw) {
                float acc = wdot(A.ff_W1 + ((size_t)l * kFF + r) * kHS, sh, kHS);
                if (lane == 0) A.ffa[r] = fmaxf(acc + A.ff_b1[(size_t)l * kFF + r], 0.0f);
            }
            gbar(A.bar);

            // P11: r3 = x2 + ff_W2 @ ffa + ff_b2
            stage_async(A.ffa, sh, kFF);
            for (int r = wid; r < kHS; r += nw) {
                float acc = wdot(A.ff_W2 + ((size_t)l * kHS + r) * kFF, sh, kFF);
                if (lane == 0) A.r3[r] = A.x2[r] + acc + A.ff_b2[(size_t)l * kHS + r];
            }
            gbar(A.bar);

            // P12: fdec = LN(r3; ln[l,2])
            if (blockIdx.x == 0)
                block_ln(A.r3, A.fdec, A.ln_g + ((size_t)l * 3 + 2) * kHS,
                                       A.ln_b + ((size_t)l * 3 + 2) * kHS, red);
            gbar(A.bar);

            // P13: LSTM cell update (block 0)
            if (blockIdx.x == 0) {
                for (int j = tid; j < kHS; j += 256) {
                    const float it = sigm(A.z[j]);
                    const float gt = tanhf(A.z[2 * kHS + j]);
                    const float cp = A.c[(size_t)l * kHS + j];
                    const float ft = sigm(cp - A.fdec[j]);
                    const float cn = fmaf(ft, cp, it * gt);
                    const float hn = tanhf(cn);
                    A.c[(size_t)l * kHS + j] = cn;
                    A.h[(size_t)l * kHS + j] = hn;
                    A.xinb[j] = hn;
                    if (l == kL - 1) A.out[(size_t)t * kHS + j] = hn;
                }
            }
            gbar(A.bar);
        }
    }

    // Final: h_T (L,1,HS) then c_T (L,1,HS) appended after hidden (T,HS)
    if (blockIdx.x == 0) {
        for (int j = tid; j < kL * kHS; j += 256) {
            A.out[(size_t)kT * kHS + j]             = A.h[j];
            A.out[(size_t)kT * kHS + kL * kHS + j]  = A.c[j];
        }
    }
}

// ---------------------------------------------------------------------------
// Host launcher
// ---------------------------------------------------------------------------
extern "C" void kernel_launch(void* const* d_in, const int* in_sizes, int n_in,
                              void* d_out, int out_size, void* d_ws, size_t ws_size,
                              hipStream_t stream) {
    const float* x       = (const float*)d_in[0];
    const float* hist    = (const float*)d_in[1];
    const float* Wx      = (const float*)d_in[2];
    const float* bx      = (const float*)d_in[3];
    const float* Wh      = (const float*)d_in[4];
    const float* bh      = (const float*)d_in[5];
    const float* sa_Win  = (const float*)d_in[6];
    const float* sa_bin  = (const float*)d_in[7];
    const float* sa_Wo   = (const float*)d_in[8];
    const float* sa_bo   = (const float*)d_in[9];
    const float* ca_Win  = (const float*)d_in[10];
    const float* ca_bin  = (const float*)d_in[11];
    const float* ca_Wo   = (const float*)d_in[12];
    const float* ca_bo   = (const float*)d_in[13];
    const float* ff_W1   = (const float*)d_in[14];
    const float* ff_b1   = (const float*)d_in[15];
    const float* ff_W2   = (const float*)d_in[16];
    const float* ff_b2   = (const float*)d_in[17];
    const float* ln_g    = (const float*)d_in[18];
    const float* ln_b    = (const float*)d_in[19];
    float* out = (float*)d_out;

    char* ws = (char*)d_ws;
    size_t off = 0;
    auto alloc = [&](size_t bytes) -> char* {
        char* p = ws + off;
        off += (bytes + 255) & ~(size_t)255;
        return p;
    };

    unsigned short* histb = (unsigned short*)alloc((size_t)kL * kM * kHS * 2);
    unsigned short* wkb   = (unsigned short*)alloc((size_t)kL * kHS * kHS * 2);
    unsigned short* wvb   = (unsigned short*)alloc((size_t)kL * kHS * kHS * 2);
    float* Kf   = (float*)alloc((size_t)kL * kM * kHS * 4);
    float* Vt   = (float*)alloc((size_t)kL * kHS * kM * 4);
    float* z    = (float*)alloc(3 * kHS * 4);
    float* f0   = (float*)alloc(kHS * 4);
    float* vsa  = (float*)alloc(kHS * 4);
    float* r1   = (float*)alloc(kHS * 4);
    float* x1   = (float*)alloc(kHS * 4);
    float* q    = (float*)alloc(kHS * 4);
    float* sc   = (float*)alloc((size_t)kNH * kM * 4);
    float* ctx  = (float*)alloc(kHS * 4);
    float* r2   = (float*)alloc(kHS * 4);
    float* x2   = (float*)alloc(kHS * 4);
    float* ffa  = (float*)alloc((size_t)kFF * 4);
    float* r3   = (float*)alloc(kHS * 4);
    float* fdec = (float*)alloc(kHS * 4);
    float* xinb = (float*)alloc(kHS * 4);
    float* hbuf = (float*)alloc((size_t)kL * kHS * 4);
    float* cbuf = (float*)alloc((size_t)kL * kHS * 4);
    int*   bar  = (int*)alloc(2 * sizeof(int));
    (void)ws_size; (void)in_sizes; (void)n_in; (void)out_size;

    // 1) bf16 packing + state init
    prep_kernel<<<4096, 256, 0, stream>>>(hist, ca_Win, histb, wkb, wvb, hbuf, cbuf, bar);

    // 2) K/V projection GEMM (WMMA bf16): 16384 tiles, one wave each, 8/block.
    kv_gemm_kernel<<<2048, 256, 0, stream>>>(histb, wkb, wvb, ca_bin, Kf, Vt);

    // 3) Persistent sequential recurrence (64 blocks spin-barriered).
    RecArgs A;
    A.x = x; A.Wx = Wx; A.bx = bx; A.Wh = Wh; A.bh = bh;
    A.sa_Win = sa_Win; A.sa_bin = sa_bin; A.sa_Wo = sa_Wo; A.sa_bo = sa_bo;
    A.ca_Win = ca_Win; A.ca_bin = ca_bin; A.ca_Wo = ca_Wo; A.ca_bo = ca_bo;
    A.ff_W1 = ff_W1; A.ff_b1 = ff_b1; A.ff_W2 = ff_W2; A.ff_b2 = ff_b2;
    A.ln_g = ln_g; A.ln_b = ln_b;
    A.Kf = Kf; A.Vt = Vt;
    A.z = z; A.f0 = f0; A.vsa = vsa; A.r1 = r1; A.x1 = x1; A.q = q; A.sc = sc;
    A.ctx = ctx; A.r2 = r2; A.x2 = x2; A.ffa = ffa; A.r3 = r3; A.fdec = fdec;
    A.xinb = xinb; A.h = hbuf; A.c = cbuf; A.out = out; A.bar = bar;
    recurrence_kernel<<<64, 256, 0, stream>>>(A);
}
